// GaussianEncoder_77584289235260
// MI455X (gfx1250) — matrix operand order repaired and checked by reading, not compiled, not run on hardware
//
#include <hip/hip_runtime.h>
#include <hip/hip_bf16.h>
#include <math.h>

typedef __attribute__((ext_vector_type(16))) _Float16 v16h;
typedef __attribute__((ext_vector_type(8)))  _Float16 v8h;
typedef __attribute__((ext_vector_type(8)))  float    v8f;

#define EMBED 128
#define NANCH 3200
#define NP 13
#define NCAM 4
#define NLV 4
#define NG 4
#define WCOLS 208   // LV*P*G

__device__ __forceinline__ float safe_sigmoid(float x) {
  x = fminf(fmaxf(x, -15.f), 15.f);
  return 1.f / (1.f + __expf(-x));
}

// ---------------------------------------------------------------------------
// Feature map transpose: [cam][128][HW] -> [cam][HW][128] (channel-last)
// ---------------------------------------------------------------------------
__global__ void transpose_fm_kernel(const float* __restrict__ in,
                                    float* __restrict__ out, int HW) {
  __shared__ float tile[32][33];
  const int c   = blockIdx.z;
  const int hw0 = blockIdx.x * 32;
  const int e0  = blockIdx.y * 32;
  const int tx = threadIdx.x, ty = threadIdx.y;
  const float* src = in  + (size_t)c * EMBED * HW;
  float*       dst = out + (size_t)c * EMBED * HW;
#pragma unroll
  for (int i = 0; i < 32; i += 8) {
    int e = e0 + ty + i, hw = hw0 + tx;
    tile[ty + i][tx] = (hw < HW) ? src[(size_t)e * HW + hw] : 0.f;
  }
  __syncthreads();
#pragma unroll
  for (int i = 0; i < 32; i += 8) {
    int hw = hw0 + ty + i;
    if (hw < HW) dst[(size_t)hw * EMBED + e0 + tx] = tile[tx][ty + i];
  }
}

// ---------------------------------------------------------------------------
// Camera MLP + camw[4][208] = cam_emb @ wfc_w + wfc_b. Single block, 128 thr.
// ---------------------------------------------------------------------------
__device__ __forceinline__ float block_sum(float v, float* red, int t) {
  red[t] = v; __syncthreads();
  for (int s = 64; s > 0; s >>= 1) { if (t < s) red[t] += red[t + s]; __syncthreads(); }
  float r = red[0]; __syncthreads();
  return r;
}

__global__ void prep_cam_kernel(const float* __restrict__ proj,
                                const float* __restrict__ w1, const float* __restrict__ b1,
                                const float* __restrict__ g1, const float* __restrict__ be1,
                                const float* __restrict__ w2, const float* __restrict__ b2,
                                const float* __restrict__ g2, const float* __restrict__ be2,
                                const float* __restrict__ wfc_w, const float* __restrict__ wfc_b,
                                float* __restrict__ camw) {
  __shared__ float sh[EMBED];
  __shared__ float red[EMBED];
  const int t = threadIdx.x;
  for (int c = 0; c < NCAM; ++c) {
    // layer 1: 12 -> 128, relu, LN
    float v = b1[t];
#pragma unroll
    for (int j = 0; j < 12; ++j)
      v += proj[c * 16 + j] * w1[j * EMBED + t];   // first 3 rows flattened
    v = fmaxf(v, 0.f);
    float m = block_sum(v, red, t) * (1.f / EMBED);
    float d = v - m;
    float var = block_sum(d * d, red, t) * (1.f / EMBED);
    v = d * rsqrtf(var + 1e-5f) * g1[t] + be1[t];
    sh[t] = v; __syncthreads();
    // layer 2: 128 -> 128, relu, LN
    float v2 = b2[t];
    for (int k = 0; k < EMBED; ++k) v2 += sh[k] * w2[k * EMBED + t];
    v2 = fmaxf(v2, 0.f);
    m = block_sum(v2, red, t) * (1.f / EMBED);
    d = v2 - m;
    var = block_sum(d * d, red, t) * (1.f / EMBED);
    v2 = d * rsqrtf(var + 1e-5f) * g2[t] + be2[t];
    __syncthreads();
    sh[t] = v2; __syncthreads();
    for (int k = t; k < WCOLS; k += EMBED) {
      float s = wfc_b[k];
      for (int e = 0; e < EMBED; ++e) s += sh[e] * wfc_w[e * WCOLS + k];
      camw[c * WCOLS + k] = s;
    }
    __syncthreads();
  }
}

// A = (instance_feature + anchor_embed) as f16, row-major [M][128]
__global__ void add_to_half_kernel(const float* __restrict__ a,
                                   const float* __restrict__ b,
                                   _Float16* __restrict__ o, int n) {
  int i = blockIdx.x * blockDim.x + threadIdx.x;
  if (i < n) o[i] = (_Float16)(a[i] + b[i]);
}

// ---------------------------------------------------------------------------
// Pack B[K=128][N] (f32) into per-lane WMMA layout (f16):
// Bp[((tn*4 + kb)*32 + lane)*16 + e] = B[kb*32 + (lane>>4)*16 + e][tn*16 + (lane&15)]
// => each lane's 16 halves for one (tile, kstep) are contiguous (two b128 loads)
// ---------------------------------------------------------------------------
__global__ void pack_b_kernel(const float* __restrict__ B,
                              _Float16* __restrict__ Bp, int N, int tilesN) {
  int i = blockIdx.x * blockDim.x + threadIdx.x;
  int total = tilesN * 4 * 32 * 16;
  if (i >= total) return;
  int e    = i & 15;
  int lane = (i >> 4) & 31;
  int kb   = (i >> 9) & 3;
  int tn   = i >> 11;
  int row  = kb * 32 + (lane >> 4) * 16 + e;
  int col  = tn * 16 + (lane & 15);
  Bp[i] = (_Float16)B[(size_t)row * N + col];
}

// ---------------------------------------------------------------------------
// f16 WMMA GEMM, K=128 fixed: one wave per 16x16 tile, fully unrolled K
// (4 x v_wmma_f32_16x16x32_f16). A loaded as 2 x b128 per K-step (contiguous
// K order k = 16*half + s, one 16-half run per lane); B pre-packed per lane.
// ---------------------------------------------------------------------------
template <bool HAS_BIAS>
__global__ void gemm_f16_kernel(const _Float16* __restrict__ A,
                                const _Float16* __restrict__ Bp,
                                float* __restrict__ C,
                                const float* __restrict__ bias,
                                int N, int tilesN) {
  const int wave = (blockIdx.x * blockDim.x + threadIdx.x) >> 5;
  const int tm = wave / tilesN, tn = wave % tilesN;
  const int lane = threadIdx.x & 31;
  const int half = lane >> 4;
  const int l16  = lane & 15;
  const int m0 = tm << 4, n0 = tn << 4;

  const _Float16* arow = A + (size_t)(m0 + l16) * 128 + half * 16;
  const _Float16* bp   = Bp + ((size_t)tn * 4 * 32 + lane) * 16;

  v8f acc = {};
#pragma unroll
  for (int kb = 0; kb < 4; ++kb) {
    v8h a0 = *(const v8h*)(arow + kb * 32);        // K = kb*32 + half*16 + 0..7
    v8h a1 = *(const v8h*)(arow + kb * 32 + 8);    // K = kb*32 + half*16 + 8..15
    v8h b0 = *(const v8h*)(bp + kb * 512);
    v8h b1 = *(const v8h*)(bp + kb * 512 + 8);
    v16h a = __builtin_shufflevector(a0, a1, 0,1,2,3,4,5,6,7,8,9,10,11,12,13,14,15);
    v16h b = __builtin_shufflevector(b0, b1, 0,1,2,3,4,5,6,7,8,9,10,11,12,13,14,15);
    acc = __builtin_amdgcn_wmma_f32_16x16x32_f16(false, a, false, b,
                                                 (short)0, acc, false, false);
  }
  float bv = HAS_BIAS ? bias[n0 + l16] : 0.f;      // single load, shared by rows
#pragma unroll
  for (int r = 0; r < 8; ++r) {                    // C layout: M = r + half*8
    C[(size_t)(m0 + r + half * 8) * N + (n0 + l16)] = acc[r] + bv;
  }
}

// ---------------------------------------------------------------------------
// Keypoint generation + projection to all cameras. One thread per anchor.
// ---------------------------------------------------------------------------
__global__ void keypoints_kernel(const float* __restrict__ IF,
                                 const float* __restrict__ anchor,
                                 const float* __restrict__ lw,  // [128][18]
                                 const float* __restrict__ lb,  // [18]
                                 const float* __restrict__ proj,// [4][4][4]
                                 const float* __restrict__ wh,  // [4][2]
                                 float* __restrict__ pts,       // [(n*13+p)*4+c][2]
                                 float* __restrict__ maskf) {   // [(n*13+p)*4+c]
  const int n = blockIdx.x * blockDim.x + threadIdx.x;
  if (n >= NANCH) return;

  float dot[18];
#pragma unroll
  for (int j = 0; j < 18; ++j) dot[j] = lb[j];
  const float* f = IF + (size_t)n * EMBED;
  for (int e = 0; e < EMBED; ++e) {
    float fe = f[e];
    const float* w = lw + e * 18;
#pragma unroll
    for (int j = 0; j < 18; ++j) dot[j] += fe * w[j];
  }

  const float FIXS[7][3] = {{0,0,0},{0.45f,0,0},{-0.45f,0,0},{0,0.45f,0},
                            {0,-0.45f,0},{0,0,0.45f},{0,0,-0.45f}};
  const float* an = anchor + (size_t)n * 10;
  float gs[3];
#pragma unroll
  for (int d2 = 0; d2 < 3; ++d2) gs[d2] = 0.01f + 3.19f * safe_sigmoid(an[3 + d2]);

  float qw = an[6], qx = an[7], qy = an[8], qz = an[9];
  float qn = sqrtf(qw*qw + qx*qx + qy*qy + qz*qz) + 1e-8f;
  qw /= qn; qx /= qn; qy /= qn; qz /= qn;
  float R[3][3];
  R[0][0] = 1 - 2*(qy*qy + qz*qz); R[0][1] = 2*(qx*qy - qw*qz); R[0][2] = 2*(qx*qz + qw*qy);
  R[1][0] = 2*(qx*qy + qw*qz); R[1][1] = 1 - 2*(qx*qx + qz*qz); R[1][2] = 2*(qy*qz - qw*qx);
  R[2][0] = 2*(qx*qz - qw*qy); R[2][1] = 2*(qy*qz + qw*qx); R[2][2] = 1 - 2*(qx*qx + qy*qy);

  float cx = safe_sigmoid(an[0]) * 40.f - 20.f;
  float cy = safe_sigmoid(an[1]) * 40.f - 20.f;
  float cz = safe_sigmoid(an[2]) * 3.2f - 2.3f;

#pragma unroll
  for (int p = 0; p < NP; ++p) {
    float s0, s1, s2;
    if (p < 7) { s0 = FIXS[p][0]; s1 = FIXS[p][1]; s2 = FIXS[p][2]; }
    else {
      int j = (p - 7) * 3;
      s0 = (safe_sigmoid(dot[j + 0]) - 0.5f) * 6.f;
      s1 = (safe_sigmoid(dot[j + 1]) - 0.5f) * 6.f;
      s2 = (safe_sigmoid(dot[j + 2]) - 0.5f) * 6.f;
    }
    float k0 = s0 * gs[0], k1 = s1 * gs[1], k2 = s2 * gs[2];
    float kx = R[0][0]*k0 + R[1][0]*k1 + R[2][0]*k2 + cx;
    float ky = R[0][1]*k0 + R[1][1]*k1 + R[2][1]*k2 + cy;
    float kz = R[0][2]*k0 + R[1][2]*k1 + R[2][2]*k2 + cz;
#pragma unroll
    for (int c = 0; c < NCAM; ++c) {
      const float* P = proj + c * 16;
      float px = P[0]*kx + P[1]*ky + P[2]*kz  + P[3];
      float py = P[4]*kx + P[5]*ky + P[6]*kz  + P[7];
      float pz = P[8]*kx + P[9]*ky + P[10]*kz + P[11];
      float dd = fmaxf(pz, 1e-5f);
      float xn = px / dd / wh[c * 2 + 0];
      float yn = py / dd / wh[c * 2 + 1];
      bool m = (pz > 1e-5f) && (xn >= 0.f) && (xn <= 1.f) && (yn >= 0.f) && (yn <= 1.f);
      int idx = (n * NP + p) * NCAM + c;
      pts[idx * 2 + 0] = xn;
      pts[idx * 2 + 1] = yn;
      maskf[idx] = m ? 1.f : 0.f;
    }
  }
}

// ---------------------------------------------------------------------------
// Fused masked softmax + deformable aggregation. One block (128 thr) / anchor;
// thread = channel, group g = tid>>5 => weights are wave-uniform scalars.
// ---------------------------------------------------------------------------
__global__ void dfa_kernel(const float* __restrict__ base,   // [N][208]
                           const float* __restrict__ camw_g, // [4][208]
                           const float* __restrict__ pts,    // [N*13*4][2]
                           const float* __restrict__ maskf,  // [N*13*4]
                           const float* __restrict__ fmT0, const float* __restrict__ fmT1,
                           const float* __restrict__ fmT2, const float* __restrict__ fmT3,
                           _Float16* __restrict__ featH) {   // [N][128]
  __shared__ float baseS[WCOLS];
  __shared__ float camwS[NCAM * WCOLS];
  __shared__ float wgtS[NP * NCAM * NLV * NG];   // 832, layout p*64+c*16+l*4+g
  __shared__ float ptsS[NP * NCAM * 2];
  __shared__ float maskS[NP * NCAM];
  __shared__ float red[EMBED];
  __shared__ float gval[NG];
  __shared__ int nvis;

  const int n = blockIdx.x;
  const int t = threadIdx.x;

  for (int i = t; i < WCOLS; i += EMBED)        baseS[i] = base[(size_t)n * WCOLS + i];
  for (int i = t; i < NCAM * WCOLS; i += EMBED) camwS[i] = camw_g[i];
  if (t < NP * NCAM * 2) ptsS[t] = pts[(size_t)n * NP * NCAM * 2 + t];
  if (t < NP * NCAM)     maskS[t] = maskf[(size_t)n * NP * NCAM + t];
  __syncthreads();
  if (t == 0) {
    int nv = 0;
    for (int j = 0; j < NP * NCAM; ++j) nv += (maskS[j] > 0.5f);
    nvis = nv;
  }

  float pmax = -3e38f;
  for (int i = t; i < 832; i += EMBED) {
    int g = i & 3, l = (i >> 2) & 3, c = (i >> 4) & 3, p = i >> 6;
    int k = l * 52 + p * 4 + g;                 // wfc column index
    float lg = (maskS[p * 4 + c] > 0.5f) ? (baseS[k] + camwS[c * WCOLS + k]) : -1e30f;
    wgtS[i] = lg;
    pmax = fmaxf(pmax, lg);
  }
  red[t] = pmax; __syncthreads();
  if (t < NG) {
    float m = -3e38f;
    for (int s = 0; s < 32; ++s) m = fmaxf(m, red[t + 4 * s]);
    gval[t] = m;
  }
  __syncthreads();
  float mg = gval[t & 3];
  float psum = 0.f;
  for (int i = t; i < 832; i += EMBED) {
    float e = __expf(wgtS[i] - mg);
    wgtS[i] = e; psum += e;
  }
  red[t] = psum; __syncthreads();
  if (t < NG) {
    float s2 = 0.f;
    for (int s = 0; s < 32; ++s) s2 += red[t + 4 * s];
    gval[t] = s2;
  }
  __syncthreads();
  float inv = (nvis > 0) ? 1.f / gval[t & 3] : 0.f;
  for (int i = t; i < 832; i += EMBED) wgtS[i] *= inv;
  __syncthreads();

  const float* fms[4] = {fmT0, fmT1, fmT2, fmT3};
  const int Hs[4] = {64, 32, 16, 8};
  const int Wd[4] = {176, 88, 44, 22};
  const int g = t >> 5;
  float acc = 0.f;

  for (int p = 0; p < NP; ++p)
    for (int c = 0; c < NCAM; ++c) {
      if (maskS[p * 4 + c] < 0.5f) continue;    // block-uniform branch
      float xn = ptsS[(p * 4 + c) * 2 + 0];
      float yn = ptsS[(p * 4 + c) * 2 + 1];
#pragma unroll
      for (int l = 0; l < NLV; ++l) {
        float wl = wgtS[p * 64 + c * 16 + l * 4 + g];   // wave-uniform
        int W = Wd[l], H = Hs[l];
        float x = xn * W - 0.5f, y = yn * H - 0.5f;
        float x0f = floorf(x), y0f = floorf(y);
        float wx = x - x0f, wy = y - y0f;
        int x0 = (int)x0f, y0 = (int)y0f;
        const float* fb = fms[l] + ((size_t)c * H * W) * EMBED + t;
        auto corner = [&](int xi, int yi) -> float {
          if (xi < 0 || xi >= W || yi < 0 || yi >= H) return 0.f;
          return fb[((size_t)yi * W + xi) * EMBED];     // 512B coalesced
        };
        float v = corner(x0, y0)         * (1.f - wx) * (1.f - wy)
                + corner(x0 + 1, y0)     * wx        * (1.f - wy)
                + corner(x0, y0 + 1)     * (1.f - wx) * wy
                + corner(x0 + 1, y0 + 1) * wx        * wy;
        acc += wl * v;
      }
    }
  featH[(size_t)n * EMBED + t] = (_Float16)acc;
}

// ---------------------------------------------------------------------------
extern "C" void kernel_launch(void* const* d_in, const int* in_sizes, int n_in,
                              void* d_out, int out_size, void* d_ws, size_t ws_size,
                              hipStream_t stream) {
  (void)in_sizes; (void)n_in; (void)out_size; (void)ws_size;
  const float* IF     = (const float*)d_in[0];
  const float* anchor = (const float*)d_in[1];
  const float* AE     = (const float*)d_in[2];
  const float* fm[4]  = {(const float*)d_in[3], (const float*)d_in[4],
                         (const float*)d_in[5], (const float*)d_in[6]};
  const float* proj   = (const float*)d_in[7];
  const float* wh     = (const float*)d_in[8];
  const float* lw     = (const float*)d_in[9];
  const float* lb     = (const float*)d_in[10];
  const float* cw1    = (const float*)d_in[11];
  const float* cb1    = (const float*)d_in[12];
  const float* cg1    = (const float*)d_in[13];
  const float* cbe1   = (const float*)d_in[14];
  const float* cw2    = (const float*)d_in[15];
  const float* cb2    = (const float*)d_in[16];
  const float* cg2    = (const float*)d_in[17];
  const float* cbe2   = (const float*)d_in[18];
  const float* wfc_w  = (const float*)d_in[19];
  const float* wfc_b  = (const float*)d_in[20];
  const float* out_w  = (const float*)d_in[21];
  const float* out_b  = (const float*)d_in[22];
  float* out = (float*)d_out;

  // workspace layout
  char* ws = (char*)d_ws;
  size_t off = 0;
  auto take = [&](size_t bytes) { size_t r = off; off += (bytes + 255) & ~(size_t)255; return r; };
  float*    camw  = (float*)(ws + take(NCAM * WCOLS * 4));
  _Float16* wfcP  = (_Float16*)(ws + take((size_t)EMBED * WCOLS * 2));   // packed B
  _Float16* outP  = (_Float16*)(ws + take((size_t)EMBED * EMBED * 2));   // packed B
  _Float16* A_h   = (_Float16*)(ws + take((size_t)NANCH * EMBED * 2));
  float*    baseW = (float*)(ws + take((size_t)NANCH * WCOLS * 4));
  float*    pts   = (float*)(ws + take((size_t)NANCH * NP * NCAM * 2 * 4));
  float*    maskf = (float*)(ws + take((size_t)NANCH * NP * NCAM * 4));
  _Float16* featH = (_Float16*)(ws + take((size_t)NANCH * EMBED * 2));
  const int HWs[4] = {64 * 176, 32 * 88, 16 * 44, 8 * 22};
  float* fmT[4];
  for (int l = 0; l < 4; ++l) fmT[l] = (float*)(ws + take((size_t)NCAM * EMBED * HWs[l] * 4));

  // 1) channel-last transpose of feature maps
  for (int l = 0; l < 4; ++l) {
    dim3 g((HWs[l] + 31) / 32, EMBED / 32, NCAM);
    transpose_fm_kernel<<<g, dim3(32, 8), 0, stream>>>(fm[l], fmT[l], HWs[l]);
  }
  // 2) camera MLP -> camw
  prep_cam_kernel<<<1, 128, 0, stream>>>(proj, cw1, cb1, cg1, cbe1, cw2, cb2,
                                         cg2, cbe2, wfc_w, wfc_b, camw);
  // 3) pack weight matrices into per-lane WMMA layout (f16)
  pack_b_kernel<<<(EMBED * WCOLS + 255) / 256, 256, 0, stream>>>(wfc_w, wfcP,
                                                                 WCOLS, WCOLS / 16);
  pack_b_kernel<<<(EMBED * EMBED + 255) / 256, 256, 0, stream>>>(out_w, outP,
                                                                 EMBED, EMBED / 16);
  // 4) A = IF + AE in f16
  add_to_half_kernel<<<(NANCH * EMBED) / 256, 256, 0, stream>>>(IF, AE, A_h,
                                                                NANCH * EMBED);
  // 5) keypoints + camera projection
  keypoints_kernel<<<(NANCH + 127) / 128, 128, 0, stream>>>(IF, anchor, lw, lb,
                                                            proj, wh, pts, maskf);
  // 6) base logits GEMM: [3200,128] x [128,208] (WMMA f16)
  {
    int tilesN = WCOLS / 16, waves = (NANCH / 16) * tilesN;   // 2600 waves
    gemm_f16_kernel<false><<<waves * 32 / 256, 256, 0, stream>>>(A_h, wfcP, baseW,
                                                                 nullptr, WCOLS, tilesN);
  }
  // 7) fused softmax + deformable aggregation
  dfa_kernel<<<NANCH, 128, 0, stream>>>(baseW, camw, pts, maskf,
                                        fmT[0], fmT[1], fmT[2], fmT[3], featH);
  // 8) output projection: [3200,128] x [128,128] + bias (WMMA f16)
  {
    int tilesN = EMBED / 16, waves = (NANCH / 16) * tilesN;   // 1600 waves
    gemm_f16_kernel<true><<<waves * 32 / 256, 256, 0, stream>>>(featH, outP, out,
                                                                out_b, EMBED, tilesN);
  }
}
